// CNV_w1a1_Balanced_Resnet_bram_23021024707488
// MI455X (gfx1250) — 8-bit matrix operand order repaired and checked by reading, compiled, not run on hardware
//
#include <hip/hip_runtime.h>
#include <cstdint>
#include <cstddef>


typedef int v8i __attribute__((ext_vector_type(8)));
typedef __attribute__((address_space(3))) int8_t* lds_i8_t;

#define EPSV 1e-4f

__global__ void k_zero_f32(float* p, int n) {
  for (int i = blockIdx.x * blockDim.x + threadIdx.x; i < n; i += gridDim.x * blockDim.x)
    p[i] = 0.f;
}

__global__ void k_input_quant(const float* __restrict__ x, int8_t* __restrict__ out, int n) {
  int i = blockIdx.x * blockDim.x + threadIdx.x;
  if (i >= n) return;
  float v = 2.f * x[i] - 1.f;
  float t = rintf(v * 128.f);
  t = fminf(fmaxf(t, -128.f), 127.f);
  out[i] = (int8_t)t;
}

__global__ void k_quant_w(const float* __restrict__ w, int8_t* __restrict__ wq,
                          int Ktot, int Kpad, int n, int wmode) {
  int i = blockIdx.x * blockDim.x + threadIdx.x;
  if (i >= n) return;
  int o = i / Kpad, k = i - o * Kpad;
  int8_t q = 0;
  if (k < Ktot) {
    float v = w[(size_t)o * Ktot + k];
    if (wmode == 0) {
      q = (v >= 0.f) ? (int8_t)1 : (int8_t)-1;
    } else {
      float t = rintf(v * 7.f);
      t = fminf(fmaxf(t, -7.f), 7.f);
      q = (int8_t)t;
    }
  }
  wq[i] = q;
}

__global__ __launch_bounds__(128)
void k_conv_wmma(const int8_t* __restrict__ act, const int8_t* __restrict__ wq,
                 float* __restrict__ out, float scale,
                 int Cin, int H, int W, int Cout, int Kdim, int pad,
                 int Ktot, int Kpad, int Mtot) {
  __shared__ alignas(64) int8_t lA[64 * 128];
  __shared__ alignas(64) int8_t lB[16 * 128];

  const int tid  = threadIdx.x;
  const int lane = tid & 31;
  const int wv   = tid >> 5;
  const int m    = lane & 15;
  const int hi   = lane >> 4;
  const int mBase = blockIdx.x * 64;
  const int nBase = blockIdx.y * 16;
  const int HW = H * W;
  const int kk = Kdim * Kdim;

  const int rowA  = tid >> 1;
  const int halfA = tid & 1;
  const int gmA   = mBase + rowA;
  const bool rowOk = (gmA < Mtot);
  int pA = 0, oyA = 0, oxA = 0, bbA = 0;
  if (rowOk) {
    bbA = gmA / HW; pA = gmA - bbA * HW;
    oyA = pA / W;   oxA = pA - oyA * W;
  }
  const int8_t* actBase = act + (size_t)bbA * Cin * HW;
  int* const dstW = (int*)(lA + rowA * 128 + halfA * 64);

  const int nnB  = tid >> 3;
  const int segB = tid & 7;
  const int gnB  = nBase + nnB;

  v8i acc = {};

  for (int kc = 0; kc < Kpad; kc += 128) {
    if (rowOk) {
      int k0 = kc + halfA * 64;
      if (kk == 1) {
        for (int j4 = 0; j4 < 16; ++j4) {
          int wrd = 0;
          for (int u = 0; u < 4; ++u) {
            int ci = k0 + j4 * 4 + u;
            int8_t v = (ci < Ktot) ? actBase[(size_t)ci * HW + pA] : (int8_t)0;
            wrd |= ((int)(uint8_t)v) << (8 * u);
          }
          dstW[j4] = wrd;
        }
      } else {
        int ci = k0 / kk;
        int r2 = k0 - ci * kk;
        int ky = r2 / Kdim, kx = r2 - ky * Kdim;
        int k = k0;
        for (int j4 = 0; j4 < 16; ++j4) {
          int wrd = 0;
          for (int u = 0; u < 4; ++u) {
            int8_t v = 0;
            if (k < Ktot) {
              int iy = oyA + ky - pad, ix = oxA + kx - pad;
              if (iy >= 0 && iy < H && ix >= 0 && ix < W)
                v = actBase[(size_t)ci * HW + iy * W + ix];
            }
            wrd |= ((int)(uint8_t)v) << (8 * u);
            ++k;
            if (++kx == Kdim) { kx = 0; if (++ky == Kdim) { ky = 0; ++ci; } }
          }
          dstW[j4] = wrd;
        }
      }
    } else {
      for (int j4 = 0; j4 < 16; ++j4) dstW[j4] = 0;
    }

    const bool fullTile = (nBase + 16 <= Cout) && (kc + 128 <= Kpad);
    if (fullTile) {
      uint64_t gaddr = (uint64_t)(uintptr_t)(wq + (size_t)gnB * Kpad + kc + segB * 16);
      lds_i8_t laddr = (lds_i8_t)(lB + nnB * 128 + segB * 16);
      asm volatile("global_load_async_to_lds_b128 %0, %1, off"
                   :: "v"(laddr), "v"(gaddr) : "memory");
    } else {
      int kb = kc + segB * 16;
      int4 v = make_int4(0, 0, 0, 0);
      if (gnB < Cout && kb + 16 <= Kpad)
        v = *(const int4*)(wq + (size_t)gnB * Kpad + kb);
      *(int4*)(lB + nnB * 128 + segB * 16) = v;
    }
    asm volatile("s_wait_asynccnt 0x0" ::: "memory");
    __syncthreads();

#pragma unroll
    for (int s = 0; s < 2; ++s) {
      const int8_t* ar = lA + (wv * 16 + m) * 128 + s * 64;
      const int8_t* br = lB + m * 128 + s * 64;
      v8i a, b;
      {
        int4 t0 = *(const int4*)(ar + 16 * hi);
        int4 t1 = *(const int4*)(ar + 32 + 16 * hi);
        a[0] = t0.x; a[1] = t0.y; a[2] = t0.z; a[3] = t0.w;
        a[4] = t1.x; a[5] = t1.y; a[6] = t1.z; a[7] = t1.w;
      }
      {
        int4 u0 = *(const int4*)(br + 16 * hi);
        int4 u1 = *(const int4*)(br + 32 + 16 * hi);
        b[0] = u0.x; b[1] = u0.y; b[2] = u0.z; b[3] = u0.w;
        b[4] = u1.x; b[5] = u1.y; b[6] = u1.z; b[7] = u1.w;
      }
      acc = __builtin_amdgcn_wmma_i32_16x16x64_iu8(true, a, true, b, acc, false, false);
    }
    __syncthreads();
  }

  int gn = nBase + m;
  if (gn < Cout) {
    for (int r = 0; r < 8; ++r) {
      int gm = mBase + wv * 16 + r + 8 * hi;
      if (gm < Mtot) {
        int bb = gm / HW, p = gm - bb * HW;
        out[((size_t)(bb * Cout + gn)) * HW + p] = (float)acc[r] * scale;
      }
    }
  }
}

__global__ void k_bn_stats(const float* __restrict__ y, float* __restrict__ stats,
                           int C, int HW, int B) {
  int c = blockIdx.x;
  float s = 0.f, ss = 0.f;
  int stride = gridDim.y * blockDim.x;
  for (int p = blockIdx.y * blockDim.x + threadIdx.x; p < HW; p += stride) {
    for (int bb = 0; bb < B; ++bb) {
      float v = y[((size_t)(bb * C + c)) * HW + p];
      s += v; ss += v * v;
    }
  }
  __shared__ float rs[256], rq[256];
  rs[threadIdx.x] = s; rq[threadIdx.x] = ss;
  __syncthreads();
  for (int st = 128; st > 0; st >>= 1) {
    if (threadIdx.x < st) { rs[threadIdx.x] += rs[threadIdx.x + st]; rq[threadIdx.x] += rq[threadIdx.x + st]; }
    __syncthreads();
  }
  if (threadIdx.x == 0) {
    atomicAdd(&stats[2 * c], rs[0]);
    atomicAdd(&stats[2 * c + 1], rq[0]);
  }
}

__global__ void k_bn_quant(const float* __restrict__ y, const float* __restrict__ stats,
                           const float* __restrict__ g, const float* __restrict__ be,
                           int8_t* __restrict__ out, int C, int HW, float cnt, int n, int amode) {
  int i = blockIdx.x * blockDim.x + threadIdx.x;
  if (i >= n) return;
  int c = (i / HW) % C;
  float mean = stats[2 * c] / cnt;
  float var  = stats[2 * c + 1] / cnt - mean * mean;
  float v = (y[i] - mean) * rsqrtf(var + EPSV) * g[c] + be[c];
  int8_t q;
  if (amode == 0) q = (v >= 0.f) ? (int8_t)1 : (int8_t)-1;
  else { float t = rintf(v * 8.f); t = fminf(fmaxf(t, -8.f), 7.f); q = (int8_t)t; }
  out[i] = q;
}

__global__ void k_maxpool2(const int8_t* __restrict__ in, int8_t* __restrict__ out,
                           int H, int W, int n) {
  int i = blockIdx.x * blockDim.x + threadIdx.x;
  if (i >= n) return;
  int Wo = W >> 1, Ho = H >> 1;
  int x = i % Wo; int t = i / Wo; int y = t % Ho; t /= Ho;
  const int8_t* p = in + (size_t)t * H * W + (size_t)(2 * y) * W + 2 * x;
  int8_t a = p[0] > p[1] ? p[0] : p[1];
  int8_t b = p[W] > p[W + 1] ? p[W] : p[W + 1];
  out[i] = a > b ? a : b;
}

__global__ void k_add_sign(const int8_t* __restrict__ x, const int8_t* __restrict__ r,
                           int8_t* __restrict__ out, int n) {
  int i = blockIdx.x * blockDim.x + threadIdx.x;
  if (i >= n) return;
  out[i] = ((int)x[i] + (int)r[i] >= 0) ? (int8_t)1 : (int8_t)-1;
}

__global__ void k_avgpool7(const int8_t* __restrict__ in, float* __restrict__ out, int n) {
  int i = blockIdx.x * blockDim.x + threadIdx.x;
  if (i >= n) return;
  const int8_t* p = in + (size_t)i * 49;
  int s = 0;
  for (int k = 0; k < 49; ++k) s += p[k];
  out[i] = (float)s / 49.f;
}

__global__ void k_absmax_pow2(const float* __restrict__ w, int n, float* __restrict__ sout) {
  __shared__ float red[256];
  float mx = 0.f;
  for (int i = threadIdx.x; i < n; i += 256) mx = fmaxf(mx, fabsf(w[i]));
  red[threadIdx.x] = mx;
  __syncthreads();
  for (int st = 128; st > 0; st >>= 1) {
    if (threadIdx.x < st) red[threadIdx.x] = fmaxf(red[threadIdx.x], red[threadIdx.x + st]);
    __syncthreads();
  }
  if (threadIdx.x == 0) {
    float s0 = red[0] / 7.f;
    *sout = (s0 > 0.f) ? exp2f(ceilf(log2f(s0))) : 1.f;
  }
}

__global__ void k_fc(const float* __restrict__ x, const float* __restrict__ w,
                     const float* __restrict__ sptr, float* __restrict__ out,
                     int I, int O, int n) {
  int i = blockIdx.x * blockDim.x + threadIdx.x;
  if (i >= n) return;
  int b = i / O, o = i - b * O;
  float s = *sptr;
  float acc = 0.f;
  for (int k = 0; k < I; ++k) {
    float t = rintf(w[(size_t)o * I + k] / s);
    t = fminf(fmaxf(t, -7.f), 7.f);
    acc += x[(size_t)b * I + k] * (t * s);
  }
  out[i] = acc;
}

__global__ void k_bn1d_relu4(const float* __restrict__ y, const float* __restrict__ g,
                             const float* __restrict__ be, float* __restrict__ out, int O) {
  int o = blockIdx.x, t = threadIdx.x;
  float v = y[(size_t)t * O + o];
  __shared__ float rs[64], rq[64];
  rs[t] = v; rq[t] = v * v;
  __syncthreads();
  for (int st = 32; st > 0; st >>= 1) {
    if (t < st) { rs[t] += rs[t + st]; rq[t] += rq[t + st]; }
    __syncthreads();
  }
  float mean = rs[0] / 64.f, var = rq[0] / 64.f - mean * mean;
  float z = (v - mean) * rsqrtf(var + EPSV) * g[o] + be[o];
  z = fmaxf(z, 0.f);
  out[(size_t)t * O + o] = fminf(rintf(z * 8.f), 15.f) * 0.125f;
}

__global__ void k_fc3_tnorm(const float* __restrict__ x, const float* __restrict__ w,
                            const float* __restrict__ sptr, const float* __restrict__ tnw,
                            const float* __restrict__ tnb, float* __restrict__ out) {
  int t = threadIdx.x;
  int b = t >> 1, o = t & 1;
  float s = *sptr;
  float acc = 0.f;
  for (int k = 0; k < 64; ++k) {
    float q = rintf(w[o * 64 + k] / s);
    q = fminf(fmaxf(q, -7.f), 7.f);
    acc += x[b * 64 + k] * (q * s);
  }
  __shared__ float rs[128], rq[128];
  rs[t] = acc; rq[t] = acc * acc;
  __syncthreads();
  for (int st = 64; st > 0; st >>= 1) {
    if (t < st) { rs[t] += rs[t + st]; rq[t] += rq[t + st]; }
    __syncthreads();
  }
  float mean = rs[0] / 128.f, var = rq[0] / 128.f - mean * mean;
  out[t] = (acc - mean) * rsqrtf(var + EPSV) * (*tnw) + (*tnb);
}


extern "C" void kernel_launch(void* const* d_in, const int* in_sizes, int n_in,
                              void* d_out, int out_size, void* d_ws, size_t ws_size,
                              hipStream_t stream) {
  (void)in_sizes; (void)n_in; (void)out_size; (void)ws_size;
  const int B = 64;
  auto P = [&](int i) { return (const float*)d_in[i]; };
  const float* X = P(0);

  size_t off = 0;
  auto carve = [&](size_t bytes) {
    void* p = (char*)d_ws + off;
    off = (off + bytes + 255) & ~(size_t)255;
    return p;
  };
  float*  F   = (float*)carve((size_t)25690112 * 4);
  int8_t* A0  = (int8_t*)carve(9633792);
  int8_t* A1  = (int8_t*)carve(25690112);
  int8_t* A2  = (int8_t*)carve(6291456);
  int8_t* A3  = (int8_t*)carve(6291456);
  int8_t* WQ  = (int8_t*)carve(1048576);
  float*  stats = (float*)carve(2048);
  float*  xfc = (float*)carve(64 * 240 * 4);
  float*  h1  = (float*)carve(64 * 120 * 4);
  float*  h2  = (float*)carve(64 * 64 * 4);
  float*  sc  = (float*)carve(4 * 4);

  int8_t* wqc = WQ;

  auto conv_stage = [&](const float* w, int Cout, int Cin, int Kdim, int pad,
                        int H, int W, const float* g, const float* be,
                        const int8_t* inAct, int8_t* outAct,
                        float scale, int wmode, int amode) {
    int Ktot = Cin * Kdim * Kdim;
    int Kpad = ((Ktot + 63) / 64) * 64;
    int8_t* wq = wqc;
    wqc += (size_t)Cout * Kpad;
    int nW = Cout * Kpad;
    k_quant_w<<<(nW + 255) / 256, 256, 0, stream>>>(w, wq, Ktot, Kpad, nW, wmode);
    k_zero_f32<<<1, 256, 0, stream>>>(stats, 2 * Cout);
    int HW = H * W, Mtot = B * HW;
    dim3 gconv((Mtot + 63) / 64, (Cout + 15) / 16);
    k_conv_wmma<<<gconv, 128, 0, stream>>>(inAct, wq, F, scale, Cin, H, W, Cout,
                                           Kdim, pad, Ktot, Kpad, Mtot);
    k_bn_stats<<<dim3(Cout, 32), 256, 0, stream>>>(F, stats, Cout, HW, B);
    int nE = B * Cout * HW;
    k_bn_quant<<<(nE + 255) / 256, 256, 0, stream>>>(F, stats, g, be, outAct,
                                                     Cout, HW, (float)(B * HW), nE, amode);
  };
  auto pool = [&](const int8_t* in, int8_t* out, int C, int H, int W) {
    int n = B * C * (H / 2) * (W / 2);
    k_maxpool2<<<(n + 255) / 256, 256, 0, stream>>>(in, out, H, W, n);
  };

  {
    int n = B * 3 * 224 * 224;
    k_input_quant<<<(n + 255) / 256, 256, 0, stream>>>(X, A0, n);
  }
  conv_stage(P(62), 8, 3, 3, 1, 224, 224, P(58), P(3), A0, A1, (1.f / 128.f) / 7.f, 1, 1);
  pool(A1, A0, 8, 224, 224);
  conv_stage(P(63), 14, 8, 3, 1, 112, 112, P(59), P(4), A0, A1, 0.125f / 7.f, 1, 1);
  pool(A1, A0, 14, 112, 112);

  const int ics[4] = {14, 28, 60, 120};
  const int dns[4] = {16, 30, 60, 120};
  const int ups[4] = {28, 60, 120, 240};
  const int HHs[4] = {56, 28, 14, 7};
  for (int k = 0; k < 4; ++k) {
    int bb = 5 + 12 * k;
    int H = HHs[k], ic = ics[k], dn = dns[k], up = ups[k];
    float sA = (k == 0) ? 0.125f : 1.f;
    conv_stage(P(bb + 8),  dn, ic, 1, 0, H, H, P(bb + 4), P(bb + 0), A0, A1, sA, 0, 0);
    conv_stage(P(bb + 9),  up, dn, 3, 1, H, H, P(bb + 5), P(bb + 1), A1, A2, 1.f, 0, 0);
    conv_stage(P(bb + 10), dn, up, 1, 0, H, H, P(bb + 6), P(bb + 2), A2, A1, 1.f, 0, 0);
    conv_stage(P(bb + 11), up, dn, 3, 1, H, H, P(bb + 7), P(bb + 3), A1, A3, 1.f, 0, 0);
    int n = B * up * H * H;
    k_add_sign<<<(n + 255) / 256, 256, 0, stream>>>(A2, A3, A1, n);
    if (k < 3) pool(A1, A0, up, H, H);
  }

  {
    int n = B * 240;
    k_avgpool7<<<(n + 255) / 256, 256, 0, stream>>>(A1, xfc, n);
  }
  k_absmax_pow2<<<1, 256, 0, stream>>>(P(53), 120 * 240, sc + 0);
  k_fc<<<(B * 120 + 255) / 256, 256, 0, stream>>>(xfc, P(53), sc + 0, F, 240, 120, B * 120);
  k_bn1d_relu4<<<120, 64, 0, stream>>>(F, P(56), P(1), h1, 120);
  k_absmax_pow2<<<1, 256, 0, stream>>>(P(54), 64 * 120, sc + 1);
  k_fc<<<(B * 64 + 255) / 256, 256, 0, stream>>>(h1, P(54), sc + 1, F, 120, 64, B * 64);
  k_bn1d_relu4<<<64, 64, 0, stream>>>(F, P(57), P(2), h2, 64);
  k_absmax_pow2<<<1, 256, 0, stream>>>(P(55), 2 * 64, sc + 2);
  k_fc3_tnorm<<<1, 128, 0, stream>>>(h2, P(55), sc + 2, P(61), P(60), (float*)d_out);
}
